// FGW_Layer_60662118089057
// MI455X (gfx1250) — matrix operand order repaired and checked by reading, not compiled, not run on hardware
//
#include <hip/hip_runtime.h>

typedef __attribute__((ext_vector_type(16))) _Float16 v16h;
typedef __attribute__((ext_vector_type(8)))  float    v8f;

#define B_GRAPHS 8192
#define N_NODES  10
#define P_PROTO  10
#define D_FEAT   128
#define K_PROTO  64
#define A_DIM    32
#define NALL     200000
#define EPS_INV  20.0f      // 1/0.05
#define TINYV    1e-9f

// workspace layout (bytes)
#define WS_ZPROTO 0        // 10*32 f32   = 1280
#define WS_C2SQ   1280     // 640 f32     = 2560
#define WS_FSQ    3840     // 640 f32     = 2560
#define WS_C2     6400     // 6400 f32    = 25600
#define WS_BFRAG  32000    // 81920 f16   = 163840 (32B aligned)

__device__ __forceinline__ float wave_sum32(float v) {
#pragma unroll
    for (int off = 16; off > 0; off >>= 1) v += __shfl_xor(v, off, 32);
    return v;
}

// ---------------------------------------------------------------------------
// Prep kernel 1 (1 block): C2 (relu+symmetrize), c2sq = (1/P)*sum_q C2^2,
// z_proto = MLP(mean_k F_filter)
// ---------------------------------------------------------------------------
__global__ __launch_bounds__(256) void fgw_prep_small(
    const float* __restrict__ C_ut, const float* __restrict__ F_filter,
    const float* __restrict__ ap_w1, const float* __restrict__ ap_b1,
    const float* __restrict__ ap_w2, const float* __restrict__ ap_b2,
    float* __restrict__ zproto_g, float* __restrict__ c2sq_g,
    float* __restrict__ C2_g)
{
    __shared__ float C2_l[K_PROTO][100];
    __shared__ float fmean[P_PROTO][D_FEAT];
    __shared__ float hh[P_PROTO][A_DIM];
    const int tid = threadIdx.x;

    for (int o = tid; o < K_PROTO * 100; o += 256) (&C2_l[0][0])[o] = 0.0f;
    for (int o = tid; o < P_PROTO * D_FEAT; o += 256) {
        int p = o / D_FEAT, d = o % D_FEAT;
        float s = 0.0f;
        for (int k = 0; k < K_PROTO; ++k)
            s += F_filter[(size_t)(p * D_FEAT + d) * K_PROTO + k];
        (&fmean[0][0])[o] = s * (1.0f / K_PROTO);
    }
    __syncthreads();

    // scatter relu(C_ut) into upper+lower triangle (45 pairs x 64 k)
    for (int t = tid; t < 45 * K_PROTO; t += 256) {
        int l = t >> 6, k = t & 63;
        int i = 0, rem = l;
        while (rem >= 9 - i) { rem -= 9 - i; ++i; }
        int j = i + 1 + rem;
        float v = C_ut[l * K_PROTO + k];
        v = v > 0.0f ? v : 0.0f;
        C2_l[k][i * 10 + j] = v;
        C2_l[k][j * 10 + i] = v;
    }
    // proto MLP layer 1 (reads fmean, synced above)
    for (int o = tid; o < P_PROTO * A_DIM; o += 256) {
        int p = o >> 5, a = o & 31;
        float acc = ap_b1[a];
        for (int d = 0; d < D_FEAT; ++d) acc += fmean[p][d] * ap_w1[d * A_DIM + a];
        hh[p][a] = acc > 0.0f ? acc : 0.0f;
    }
    __syncthreads();
    for (int o = tid; o < P_PROTO * A_DIM; o += 256) {
        int p = o >> 5, a = o & 31;
        float acc = ap_b2[a];
        for (int j = 0; j < A_DIM; ++j) acc += hh[p][j] * ap_w2[j * A_DIM + a];
        zproto_g[o] = acc;
    }
    for (int o = tid; o < K_PROTO * P_PROTO; o += 256) {
        int k = o / 10, p = o % 10;
        float s = 0.0f;
        for (int q = 0; q < 10; ++q) { float c = C2_l[k][p * 10 + q]; s += c * c; }
        c2sq_g[o] = 0.1f * s;     // q-marginal is uniform 1/P
    }
    for (int o = tid; o < K_PROTO * 100; o += 256) C2_g[o] = (&C2_l[0][0])[o];
}

// ---------------------------------------------------------------------------
// Prep kernel 2: B-matrix WMMA fragments (f16) for the x @ Ff^T GEMM, and
// fsq[k,p] = |Ff[k,p]|^2.  B fragment layout (V_WMMA_F32_16X16X32_F16,
// 32x16 B, v16h per lane): col n = lane&15; K = (lane<16 ? 0 : 8) + (e<8 ? e : 8+e).
// ---------------------------------------------------------------------------
__global__ __launch_bounds__(256) void fgw_prep_bfrag(
    const float* __restrict__ F_filter, _Float16* __restrict__ Bfrag_g,
    float* __restrict__ fsq_g)
{
    int g = blockIdx.x * 256 + threadIdx.x;
    int nthr = gridDim.x * 256;
    for (int o = g; o < 40 * 4 * 32 * 16; o += nthr) {
        int e = o & 15, L = (o >> 4) & 31, s = (o >> 9) & 3, t = o >> 11;
        int c = t * 16 + (L & 15);          // global column = k*10 + p (640 total)
        int k = c / 10, p = c - k * 10;
        int kk = ((L < 16) ? 0 : 8) + ((e < 8) ? e : 8 + e);   // K within the 32-chunk
        int d = s * 32 + kk;
        Bfrag_g[o] = (_Float16)F_filter[(size_t)(p * D_FEAT + d) * K_PROTO + k];
    }
    for (int o = g; o < K_PROTO * P_PROTO; o += nthr) {
        int k = o / 10, p = o - k * 10;
        float s = 0.0f;
        for (int d = 0; d < D_FEAT; ++d) {
            float f = F_filter[(size_t)(p * D_FEAT + d) * K_PROTO + k];
            s += f * f;
        }
        fsq_g[o] = s;
    }
}

// ---------------------------------------------------------------------------
// Main kernel: one block (8 waves) per graph b.
// ---------------------------------------------------------------------------
__global__ __launch_bounds__(256) void fgw_main(
    const float* __restrict__ adj, const float* __restrict__ features,
    const int* __restrict__ idxs, const float* __restrict__ ln_g,
    const float* __restrict__ ln_b, const float* __restrict__ alpha_raw,
    const float* __restrict__ an_w1, const float* __restrict__ an_b1,
    const float* __restrict__ an_w2, const float* __restrict__ an_b2,
    const float* __restrict__ gamma_p,
    const float* __restrict__ zproto_g, const float* __restrict__ c2sq_g,
    const float* __restrict__ fsq_g, const float* __restrict__ C2_g,
    const _Float16* __restrict__ Bfrag_g, float* __restrict__ out)
{
    __shared__ float x_s[16][D_FEAT];
    __shared__ __attribute__((aligned(32))) _Float16 afrag[4][32][16];
    __shared__ float M_s[K_PROTO][100];
    __shared__ float zh_s[N_NODES][A_DIM];
    __shared__ float zn_s[N_NODES][A_DIM];
    __shared__ float alpha_s[100];
    __shared__ float adj_s[100];
    __shared__ float r_s[N_NODES];
    __shared__ float p_s[16];
    __shared__ float xsq_s[16];
    __shared__ int   idx_s[N_NODES];
    __shared__ float psum_s;
    // per-wave FGW scratch
    __shared__ float Cw[8][100], Tw[8][100], Kw[8][100], tmpw[8][100];
    __shared__ float ccw[8][10], uw[8][10], vw[8][10];

    const int b = blockIdx.x;
    const int tid = threadIdx.x;
    const int lane = tid & 31;
    const int w = tid >> 5;

    if (tid < 16) { p_s[tid] = 0.0f; xsq_s[tid] = 0.0f; }
    if (tid < N_NODES) idx_s[tid] = idxs[(size_t)b * N_NODES + tid];
    __syncthreads();
    if (tid < N_NODES) p_s[tid] = (idx_s[tid] != NALL) ? (1.0f / N_NODES) : 0.0f;
    __syncthreads();
    if (tid == 0) {
        float s = 0.0f;
        for (int n = 0; n < N_NODES; ++n) s += p_s[n];
        psum_s = s + TINYV;
    }
    __syncthreads();
    if (tid < N_NODES) p_s[tid] = p_s[tid] / psum_s;

    // ---- gather + LayerNorm: wave w handles rows w and w+8; pad rows zeroed
    for (int r = w; r < 16; r += 8) {
        if (r < N_NODES) {
            int idx = idx_s[r];
            bool vrow = (idx != NALL);
            float4 xv = make_float4(0.f, 0.f, 0.f, 0.f);
            if (vrow)
                xv = *(const float4*)(features + (size_t)idx * D_FEAT + lane * 4);
            float mu = wave_sum32(xv.x + xv.y + xv.z + xv.w) * (1.0f / 128.0f);
            float dv[4] = {xv.x - mu, xv.y - mu, xv.z - mu, xv.w - mu};
            float var = wave_sum32(dv[0]*dv[0] + dv[1]*dv[1] + dv[2]*dv[2] + dv[3]*dv[3])
                        * (1.0f / 128.0f);
            float inv = 1.0f / sqrtf(var + 1e-5f);
            float q = 0.0f;
#pragma unroll
            for (int j = 0; j < 4; ++j) {
                int d = lane * 4 + j;
                float xh = dv[j] * inv * ln_g[d] + ln_b[d];
                if (!vrow) xh = 0.0f;
                x_s[r][d] = xh;
                q += xh * xh;
            }
            q = wave_sum32(q);
            if (lane == 0) xsq_s[r] = q;
        } else {
#pragma unroll
            for (int j = 0; j < 4; ++j) x_s[r][lane * 4 + j] = 0.0f;
            if (lane == 0) xsq_s[r] = 0.0f;
        }
    }
    __syncthreads();

    // ---- node MLP layer 1 + adj load
    for (int o = tid; o < N_NODES * A_DIM; o += 256) {
        int n = o >> 5, a = o & 31;
        float acc = an_b1[a];
        for (int d = 0; d < D_FEAT; ++d) acc += x_s[n][d] * an_w1[d * A_DIM + a];
        zh_s[n][a] = acc > 0.0f ? acc : 0.0f;
    }
    for (int o = tid; o < 100; o += 256) adj_s[o] = adj[(size_t)b * 100 + o];
    __syncthreads();
    // ---- layer 2
    for (int o = tid; o < N_NODES * A_DIM; o += 256) {
        int n = o >> 5, a = o & 31;
        float acc = an_b2[a];
#pragma unroll
        for (int j = 0; j < A_DIM; ++j) acc += zh_s[n][j] * an_w2[j * A_DIM + a];
        zn_s[n][a] = acc;
    }
    __syncthreads();
    // ---- alpha = sigmoid(z_node @ z_proto^T + alpha_raw); r_s; A fragments
    for (int o = tid; o < 100; o += 256) {
        int n = o / 10, p = o - n * 10;
        float acc = alpha_raw[0];
#pragma unroll
        for (int j = 0; j < A_DIM; ++j) acc += zn_s[n][j] * zproto_g[p * A_DIM + j];
        alpha_s[o] = 1.0f / (1.0f + expf(-acc));
    }
    if (tid < N_NODES) {
        float s = 0.0f;
        for (int m = 0; m < N_NODES; ++m) {
            float a = adj_s[tid * 10 + m];
            s += a * a * p_s[m];
        }
        r_s[tid] = s;
    }
    // A fragment layout (16x32 f16): m = L&15; i=e>>1,h=e&1;
    // K = (L<16?0:8) + (i<4 ? 2i+h : 16+2(i-4)+h)
    for (int o = tid; o < 4 * 32 * 16; o += 256) {
        int e = o & 15, L = (o >> 4) & 31, s = o >> 9;
        int m = L & 15;
        int i = e >> 1, h = e & 1;
        int base = (L < 16) ? 0 : 8;
        int kk = (i < 4) ? (base + 2 * i + h) : (base + 16 + 2 * (i - 4) + h);
        afrag[s][L][e] = (_Float16)x_s[m][s * 32 + kk];
    }
    __syncthreads();

    // ---- M[k,n,p] = |x_n|^2 + |Ff_kp|^2 - 2 x.Ff  via WMMA f16->f32
    {
        v16h af[4];
#pragma unroll
        for (int s = 0; s < 4; ++s)
            af[s] = *(const v16h*)(&afrag[s][lane][0]);
        for (int t = w; t < 40; t += 8) {
            v8f acc = {};
#pragma unroll
            for (int s = 0; s < 4; ++s) {
                v16h bf = *(const v16h*)(Bfrag_g + ((size_t)(t * 4 + s) * 32 + lane) * 16);
                acc = __builtin_amdgcn_wmma_f32_16x16x32_f16(
                    false, af[s], false, bf, (short)0, acc, false, false);
            }
            int half = (lane < 16) ? 0 : 8;
            int c = t * 16 + (lane & 15);   // c = k*10 + p, always < 640
            int k = c / 10, p = c - k * 10;
            float fs = fsq_g[c];
#pragma unroll
            for (int r = 0; r < 8; ++r) {
                int m = half + r;
                if (m < N_NODES)
                    M_s[k][m * 10 + p] = xsq_s[m] + fs - 2.0f * acc[r];
            }
        }
    }
    __syncthreads();

    // ---- FGW: wave w handles k = 8w .. 8w+7
    const float gam = fabsf(gamma_p[0]);
    for (int ki = 0; ki < 8; ++ki) {
        int k = w * 8 + ki;
        for (int o = lane; o < 100; o += 32) {
            Cw[w][o] = C2_g[k * 100 + o];
            Tw[w][o] = p_s[o / 10] * 0.1f;   // p (x) q init
        }
        if (lane < 10) ccw[w][lane] = c2sq_g[k * 10 + lane];
        __syncthreads();

        for (int it = 0; it < 2; ++it) {
            // tmp = adj @ T
            for (int o = lane; o < 100; o += 32) {
                int n = o / 10, qq = o - n * 10;
                float s = 0.0f;
#pragma unroll
                for (int m = 0; m < 10; ++m) s += adj_s[n * 10 + m] * Tw[w][m * 10 + qq];
                tmpw[w][o] = s;
            }
            __syncthreads();
            // tens = constC - 2 tmp @ C2k ; K = exp(-cost/eps)
            for (int o = lane; o < 100; o += 32) {
                int n = o / 10, p = o - n * 10;
                float s = 0.0f;
#pragma unroll
                for (int q = 0; q < 10; ++q) s += tmpw[w][n * 10 + q] * Cw[w][q * 10 + p];
                float tens = r_s[n] + ccw[w][p] - 2.0f * s;
                float al = alpha_s[o];
                float cost = (1.0f - al) * M_s[k][o] + al * tens;
                Kw[w][o] = expf(-cost * EPS_INV);
            }
            if (lane < 10) uw[w][lane] = p_s[lane];
            __syncthreads();
            // Sinkhorn
            for (int sk = 0; sk < 3; ++sk) {
                if (lane < 10) {
                    float s = 0.0f;
#pragma unroll
                    for (int n = 0; n < 10; ++n) s += Kw[w][n * 10 + lane] * uw[w][n];
                    vw[w][lane] = 0.1f / (s + TINYV);
                }
                __syncthreads();
                if (lane < 10) {
                    float s = 0.0f;
#pragma unroll
                    for (int p = 0; p < 10; ++p) s += Kw[w][lane * 10 + p] * vw[w][p];
                    uw[w][lane] = p_s[lane] / (s + TINYV);
                }
                __syncthreads();
            }
            for (int o = lane; o < 100; o += 32)
                Tw[w][o] = uw[w][o / 10] * Kw[w][o] * vw[w][o - (o / 10) * 10];
            __syncthreads();
        }

        // final cost & fgw = sum(T * cost)
        for (int o = lane; o < 100; o += 32) {
            int n = o / 10, qq = o - n * 10;
            float s = 0.0f;
#pragma unroll
            for (int m = 0; m < 10; ++m) s += adj_s[n * 10 + m] * Tw[w][m * 10 + qq];
            tmpw[w][o] = s;
        }
        __syncthreads();
        float accf = 0.0f;
        for (int o = lane; o < 100; o += 32) {
            int n = o / 10, p = o - n * 10;
            float s = 0.0f;
#pragma unroll
            for (int q = 0; q < 10; ++q) s += tmpw[w][n * 10 + q] * Cw[w][q * 10 + p];
            float tens = r_s[n] + ccw[w][p] - 2.0f * s;
            float al = alpha_s[o];
            float cost = (1.0f - al) * M_s[k][o] + al * tens;
            accf += Tw[w][o] * cost;
        }
        accf = wave_sum32(accf);
        if (lane == 0) out[(size_t)b * K_PROTO + k] = expf(-gam * accf);
        __syncthreads();
    }
}

// ---------------------------------------------------------------------------
extern "C" void kernel_launch(void* const* d_in, const int* in_sizes, int n_in,
                              void* d_out, int out_size, void* d_ws, size_t ws_size,
                              hipStream_t stream) {
    (void)in_sizes; (void)n_in; (void)out_size; (void)ws_size;
    const float* adj       = (const float*)d_in[0];
    const float* features  = (const float*)d_in[1];
    const int*   idxs      = (const int*)d_in[2];
    const float* C_ut      = (const float*)d_in[3];
    const float* F_filter  = (const float*)d_in[4];
    const float* ln_g      = (const float*)d_in[5];
    const float* ln_b      = (const float*)d_in[6];
    const float* alpha_raw = (const float*)d_in[7];
    const float* an_w1     = (const float*)d_in[8];
    const float* an_b1     = (const float*)d_in[9];
    const float* an_w2     = (const float*)d_in[10];
    const float* an_b2     = (const float*)d_in[11];
    const float* ap_w1     = (const float*)d_in[12];
    const float* ap_b1     = (const float*)d_in[13];
    const float* ap_w2     = (const float*)d_in[14];
    const float* ap_b2     = (const float*)d_in[15];
    const float* gamma_p   = (const float*)d_in[16];

    char* ws = (char*)d_ws;
    float*     zproto_g = (float*)(ws + WS_ZPROTO);
    float*     c2sq_g   = (float*)(ws + WS_C2SQ);
    float*     fsq_g    = (float*)(ws + WS_FSQ);
    float*     C2_g     = (float*)(ws + WS_C2);
    _Float16*  Bfrag_g  = (_Float16*)(ws + WS_BFRAG);

    hipLaunchKernelGGL(fgw_prep_small, dim3(1), dim3(256), 0, stream,
                       C_ut, F_filter, ap_w1, ap_b1, ap_w2, ap_b2,
                       zproto_g, c2sq_g, C2_g);
    hipLaunchKernelGGL(fgw_prep_bfrag, dim3(64), dim3(256), 0, stream,
                       F_filter, Bfrag_g, fsq_g);
    hipLaunchKernelGGL(fgw_main, dim3(B_GRAPHS), dim3(256), 0, stream,
                       adj, features, idxs, ln_g, ln_b, alpha_raw,
                       an_w1, an_b1, an_w2, an_b2, gamma_p,
                       zproto_g, c2sq_g, fsq_g, C2_g, Bfrag_g, (float*)d_out);
}
